// Attention_56624848830790
// MI455X (gfx1250) — compile-verified, matrix operand order checked by reading, not run on hardware
//
#include <hip/hip_runtime.h>
#include <hip/hip_bf16.h>

// Problem constants (fixed by the reference's setup_inputs; batch_max_len=25).
#define B_   512
#define T_   64
#define NIN_ 512
#define NH_  512
#define NC_  38
#define S_   26            // batch_max_len + 1
#define KCAT 1088          // 512 ctx + 38 onehot + 26 pad + 512 h  (34 * 32)
#define HOFF 576           // h_prev starts at element 576 (16B aligned in bf16)

typedef __attribute__((ext_vector_type(8)))  __bf16 bf16x8;
typedef __attribute__((ext_vector_type(16))) __bf16 bf16x16;
typedef __attribute__((ext_vector_type(8)))  float  v8f;

// ---------------------------------------------------------------------------
// WMMA helpers
// ---------------------------------------------------------------------------
__device__ inline v8f wmma_bf16(bf16x16 a, bf16x16 b, v8f c) {
  // (neg_a, A, neg_b, B, c_mod, C, reuse_a, reuse_b)
  return __builtin_amdgcn_wmma_f32_16x16x32_bf16(false, a, false, b,
                                                 (short)0, c, false, false);
}

// Load one lane's slice of a 16x32 bf16 fragment (A or B operand).
// Lane L holds row (L&15), K-chunks at (L>>4)*8 and 16+(L>>4)*8.
// p must already point at row_base + (L>>4)*8 for K-offset 0.
__device__ inline bf16x16 load_frag(const __bf16* p) {
  bf16x8 lo = *(const bf16x8*)p;
  bf16x8 hi = *(const bf16x8*)(p + 16);
  bf16x16 r;
#pragma unroll
  for (int i = 0; i < 8; ++i) { r[i] = lo[i]; r[i + 8] = hi[i]; }
  return r;
}

// ---------------------------------------------------------------------------
// GEMM: C[M,N] = A[M,K] * W[N,K]^T (+ bias[N]), A/W bf16 row-major, C f32.
// Block = 128 threads (4 waves), block tile 64x64, wave tile 32x32 (2x2 WMMA).
// M, N must be multiples of 64; K a multiple of 32 (true for all call sites).
// ---------------------------------------------------------------------------
__global__ void __launch_bounds__(128)
gemm_bf16_wmma(const __bf16* __restrict__ A, int lda,
               const __bf16* __restrict__ W, int ldw,
               const float* __restrict__ bias,
               float* __restrict__ C, int ldc, int K)
{
  const int lane = threadIdx.x & 31;
  const int wave = threadIdx.x >> 5;
  const int m0 = blockIdx.y * 64 + (wave >> 1) * 32;
  const int n0 = blockIdx.x * 64 + (wave & 1) * 32;
  const int row  = lane & 15;
  const int koff = (lane >> 4) * 8;

  const __bf16* a0p = A + (size_t)(m0 + row)      * lda + koff;
  const __bf16* a1p = A + (size_t)(m0 + 16 + row) * lda + koff;
  const __bf16* b0p = W + (size_t)(n0 + row)      * ldw + koff;
  const __bf16* b1p = W + (size_t)(n0 + 16 + row) * ldw + koff;

  v8f acc00 = {}, acc01 = {}, acc10 = {}, acc11 = {};
  for (int k = 0; k < K; k += 32) {
    bf16x16 a0 = load_frag(a0p + k);
    bf16x16 a1 = load_frag(a1p + k);
    bf16x16 b0 = load_frag(b0p + k);
    bf16x16 b1 = load_frag(b1p + k);
    // gfx1250 global_prefetch_b8 on the next K-chunks (speculative-safe).
    __builtin_prefetch(a0p + k + 64, 0, 1);
    __builtin_prefetch(a1p + k + 64, 0, 1);
    __builtin_prefetch(b0p + k + 64, 0, 1);
    __builtin_prefetch(b1p + k + 64, 0, 1);
    acc00 = wmma_bf16(a0, b0, acc00);
    acc01 = wmma_bf16(a0, b1, acc01);
    acc10 = wmma_bf16(a1, b0, acc10);
    acc11 = wmma_bf16(a1, b1, acc11);
  }

  // C/D layout: VGPR r -> M = r + (lane>>4)*8, N = lane&15.
  const int nlo   = n0 + (lane & 15);
  const int mbase = (lane >> 4) * 8;
  const float bv0 = bias ? bias[nlo]      : 0.0f;
  const float bv1 = bias ? bias[nlo + 16] : 0.0f;
#pragma unroll
  for (int r = 0; r < 8; ++r) {
    const int m = m0 + mbase + r;
    C[(size_t)m        * ldc + nlo]      = acc00[r] + bv0;
    C[(size_t)m        * ldc + nlo + 16] = acc01[r] + bv1;
    C[(size_t)(m + 16) * ldc + nlo]      = acc10[r] + bv0;
    C[(size_t)(m + 16) * ldc + nlo + 16] = acc11[r] + bv1;
  }
}

// ---------------------------------------------------------------------------
// Utility kernels
// ---------------------------------------------------------------------------
__global__ void f2bf_kernel(const float* __restrict__ in, __bf16* __restrict__ out, int n) {
  int i = blockIdx.x * blockDim.x + threadIdx.x;
  if (i < n) out[i] = (__bf16)in[i];
}

__global__ void zero_bf16_kernel(__bf16* p, int n) {
  int i = blockIdx.x * blockDim.x + threadIdx.x;
  if (i < n) p[i] = (__bf16)0.0f;
}

__global__ void zero_f32_kernel(float* p, int n) {
  int i = blockIdx.x * blockDim.x + threadIdx.x;
  if (i < n) p[i] = 0.0f;
}

// Pack [Wih | 0 | Whh] -> bf16 Wcat[2048, KCAT]; bcat = bih + bhh.
__global__ void pack_wcat_kernel(const float* __restrict__ Wih,
                                 const float* __restrict__ Whh,
                                 const float* __restrict__ bih,
                                 const float* __restrict__ bhh,
                                 __bf16* __restrict__ Wcat,
                                 float* __restrict__ bcat)
{
  int n = blockIdx.x;  // 0..4*NH-1
  for (int k = threadIdx.x; k < KCAT; k += blockDim.x) {
    float v;
    if (k < NIN_ + NC_)      v = Wih[(size_t)n * (NIN_ + NC_) + k];
    else if (k < HOFF)       v = 0.0f;
    else                     v = Whh[(size_t)n * NH_ + (k - HOFF)];
    Wcat[(size_t)n * KCAT + k] = (__bf16)v;
  }
  if (threadIdx.x == 0) bcat[n] = bih[n] + bhh[n];
}

// ---------------------------------------------------------------------------
// Attention step: e = tanh(Hproj + hp) . score ; softmax over T ;
// context = alpha . batch_hidden ; write [context | onehot] (bf16) into xh.
// One block per batch element, 256 threads (8 waves).
// ---------------------------------------------------------------------------
__global__ void __launch_bounds__(256)
attention_step_kernel(const float* __restrict__ Hproj,  // [B,T,NH]
                      const float* __restrict__ hp,     // [B,NH]
                      const float* __restrict__ BH,     // [B,T,NIN] f32
                      const float* __restrict__ score,  // [NH]
                      const int*   __restrict__ text,   // [B,26]
                      __bf16* __restrict__ xh,          // [B,KCAT]
                      int s)
{
  const int b    = blockIdx.x;
  const int tid  = threadIdx.x;
  const int lane = tid & 31;
  const int wave = tid >> 5;

  __shared__ float sh_hp[NH_];
  __shared__ float sh_sc[NH_];
  __shared__ float e_sm[T_];

  for (int i = tid; i < NH_; i += 256) {
    sh_hp[i] = hp[(size_t)b * NH_ + i];
    sh_sc[i] = score[i];
  }
  __syncthreads();

  // e[t]: one wave per t, lanes stride the NH reduction.
  for (int t = wave; t < T_; t += 8) {
    const float* hrow = Hproj + ((size_t)b * T_ + t) * NH_;
    float sum = 0.0f;
    for (int h = lane; h < NH_; h += 32)
      sum += tanhf(hrow[h] + sh_hp[h]) * sh_sc[h];
#pragma unroll
    for (int off = 16; off > 0; off >>= 1)
      sum += __shfl_xor(sum, off, 32);
    if (lane == 0) e_sm[t] = sum;
  }
  __syncthreads();

  // Softmax over T=64 with wave 0 (2 elements per lane).
  if (wave == 0) {
    float v0 = e_sm[lane], v1 = e_sm[lane + 32];
    float m = fmaxf(v0, v1);
#pragma unroll
    for (int off = 16; off > 0; off >>= 1)
      m = fmaxf(m, __shfl_xor(m, off, 32));
    float e0 = __expf(v0 - m), e1 = __expf(v1 - m);
    float ss = e0 + e1;
#pragma unroll
    for (int off = 16; off > 0; off >>= 1)
      ss += __shfl_xor(ss, off, 32);
    float inv = 1.0f / ss;
    e_sm[lane] = e0 * inv;
    e_sm[lane + 32] = e1 * inv;
  }
  __syncthreads();

  // context[d] = sum_t alpha[t] * BH[b,t,d]; each thread handles d, d+256.
  const float* brow = BH + (size_t)b * T_ * NIN_;
  float c0 = 0.0f, c1 = 0.0f;
  for (int t = 0; t < T_; ++t) {
    const float a = e_sm[t];
    c0 += a * brow[(size_t)t * NIN_ + tid];
    c1 += a * brow[(size_t)t * NIN_ + tid + 256];
  }
  __bf16* xrow = xh + (size_t)b * KCAT;
  xrow[tid]       = (__bf16)c0;
  xrow[tid + 256] = (__bf16)c1;

  if (tid < NC_) xrow[NIN_ + tid] = (__bf16)0.0f;
  __syncthreads();
  if (tid == 0) {
    int cls = text[b * 26 + s];
    xrow[NIN_ + cls] = (__bf16)1.0f;
  }
}

// ---------------------------------------------------------------------------
// LSTM pointwise: gate order i,f,g,o. Writes c (f32), h into h_all (f32) and
// into the h-section of xh (bf16) for the next step's GEMMs.
// ---------------------------------------------------------------------------
__global__ void __launch_bounds__(256)
lstm_point_kernel(const float* __restrict__ gates,  // [B, 4*NH]
                  float* __restrict__ c_state,      // [B, NH]
                  float* __restrict__ h_all,        // [B, S, NH]
                  __bf16* __restrict__ xh,          // [B, KCAT]
                  int s)
{
  int idx = blockIdx.x * blockDim.x + threadIdx.x;   // 0 .. B*NH-1
  if (idx >= B_ * NH_) return;
  int b = idx >> 9;          // /NH_
  int j = idx & (NH_ - 1);
  const float* g = gates + (size_t)b * 4 * NH_;
  float gi = g[j], gf = g[NH_ + j], gg = g[2 * NH_ + j], go = g[3 * NH_ + j];
  float si = 1.0f / (1.0f + __expf(-gi));
  float sf = 1.0f / (1.0f + __expf(-gf));
  float so = 1.0f / (1.0f + __expf(-go));
  float c  = sf * c_state[idx] + si * tanhf(gg);
  float h  = so * tanhf(c);
  c_state[idx] = c;
  h_all[((size_t)b * S_ + s) * NH_ + j] = h;
  xh[(size_t)b * KCAT + HOFF + j] = (__bf16)h;
}

// ---------------------------------------------------------------------------
// Generator: probs[b,s,c] = h_all[b,s,:] . Wgen[c,:] + bgen[c]  (tiny GEMM)
// ---------------------------------------------------------------------------
__global__ void __launch_bounds__(256)
generator_kernel(const float* __restrict__ h_all,   // [B*S, NH]
                 const float* __restrict__ Wgen,    // [NC, NH]
                 const float* __restrict__ bgen,    // [NC]
                 float* __restrict__ out)           // [B*S, NC]
{
  int idx = blockIdx.x * blockDim.x + threadIdx.x;
  if (idx >= B_ * S_ * NC_) return;
  int c  = idx % NC_;
  int bs = idx / NC_;
  const float* h = h_all + (size_t)bs * NH_;
  const float* w = Wgen  + (size_t)c * NH_;
  float acc = bgen[c];
  for (int j = 0; j < NH_; ++j) acc += h[j] * w[j];
  out[idx] = acc;
}

// ---------------------------------------------------------------------------
// Host: orchestrate the 26-step recurrence. All launches on `stream`.
// Workspace layout (all 256B aligned; ~141 MB total required).
// ---------------------------------------------------------------------------
static inline size_t align256(size_t x) { return (x + 255) & ~(size_t)255; }

extern "C" void kernel_launch(void* const* d_in, const int* in_sizes, int n_in,
                              void* d_out, int out_size, void* d_ws, size_t ws_size,
                              hipStream_t stream)
{
  const float* BH     = (const float*)d_in[0];   // [B,T,NIN]
  const int*   text   = (const int*)  d_in[1];   // [B,26]
  // d_in[2] = batch_max_len (fixed 25 in setup; S_ = 26 hardcoded)
  const float* Wi2h   = (const float*)d_in[3];   // [NH,NIN]
  const float* Wh2h   = (const float*)d_in[4];   // [NH,NH]
  const float* bh2h   = (const float*)d_in[5];   // [NH]
  const float* Wscore = (const float*)d_in[6];   // [1,NH]
  const float* Wih    = (const float*)d_in[7];   // [4NH, NIN+NC]
  const float* Whh    = (const float*)d_in[8];   // [4NH, NH]
  const float* bih    = (const float*)d_in[9];
  const float* bhh    = (const float*)d_in[10];
  const float* Wgen   = (const float*)d_in[11];  // [NC,NH]
  const float* bgen   = (const float*)d_in[12];
  float* out = (float*)d_out;

  char* ws = (char*)d_ws;
  size_t off = 0;
  __bf16* BH16   = (__bf16*)(ws + off); off = align256(off + (size_t)B_*T_*NIN_*2);
  __bf16* WI2H16 = (__bf16*)(ws + off); off = align256(off + (size_t)NH_*NIN_*2);
  __bf16* WH2H16 = (__bf16*)(ws + off); off = align256(off + (size_t)NH_*NH_*2);
  __bf16* WCAT16 = (__bf16*)(ws + off); off = align256(off + (size_t)4*NH_*KCAT*2);
  float*  BCAT   = (float*) (ws + off); off = align256(off + (size_t)4*NH_*4);
  float*  HPROJ  = (float*) (ws + off); off = align256(off + (size_t)B_*T_*NH_*4);
  float*  HP     = (float*) (ws + off); off = align256(off + (size_t)B_*NH_*4);
  __bf16* XH     = (__bf16*)(ws + off); off = align256(off + (size_t)B_*KCAT*2);
  float*  GATES  = (float*) (ws + off); off = align256(off + (size_t)B_*4*NH_*4);
  float*  CST    = (float*) (ws + off); off = align256(off + (size_t)B_*NH_*4);
  float*  HALL   = (float*) (ws + off); off = align256(off + (size_t)B_*S_*NH_*4);
  (void)ws_size; (void)in_sizes; (void)n_in; (void)out_size;

  // --- one-time prep -------------------------------------------------------
  {
    int n = B_ * T_ * NIN_;
    f2bf_kernel<<<(n + 255) / 256, 256, 0, stream>>>(BH, BH16, n);
  }
  {
    int n = NH_ * NIN_;
    f2bf_kernel<<<(n + 255) / 256, 256, 0, stream>>>(Wi2h, WI2H16, n);
    f2bf_kernel<<<(n + 255) / 256, 256, 0, stream>>>(Wh2h, WH2H16, n);
  }
  pack_wcat_kernel<<<4 * NH_, 256, 0, stream>>>(Wih, Whh, bih, bhh, WCAT16, BCAT);
  {
    int n = B_ * KCAT;
    zero_bf16_kernel<<<(n + 255) / 256, 256, 0, stream>>>(XH, n);  // h0 = 0, pads = 0
    int m = B_ * NH_;
    zero_f32_kernel<<<(m + 255) / 256, 256, 0, stream>>>(CST, m);  // c0 = 0
  }

  // H_proj = batch_hidden @ Wi2h^T   [32768 x 512 x 512]
  gemm_bf16_wmma<<<dim3(NH_ / 64, (B_ * T_) / 64), 128, 0, stream>>>(
      BH16, NIN_, WI2H16, NIN_, nullptr, HPROJ, NH_, NIN_);

  // --- recurrence ----------------------------------------------------------
  for (int s = 0; s < S_; ++s) {
    // hp = h_prev @ Wh2h^T + bh2h   [512 x 512 x 512]; h_prev lives in xh[:,576:1088]
    gemm_bf16_wmma<<<dim3(NH_ / 64, B_ / 64), 128, 0, stream>>>(
        XH + HOFF, KCAT, WH2H16, NH_, bh2h, HP, NH_, NH_);

    attention_step_kernel<<<B_, 256, 0, stream>>>(
        HPROJ, HP, BH, Wscore, text, XH, s);

    // gates = [ctx|onehot|h] @ [Wih|Whh]^T + (bih+bhh)   [512 x 2048 x 1088]
    gemm_bf16_wmma<<<dim3((4 * NH_) / 64, B_ / 64), 128, 0, stream>>>(
        XH, KCAT, WCAT16, KCAT, BCAT, GATES, 4 * NH_, KCAT);

    lstm_point_kernel<<<(B_ * NH_ + 255) / 256, 256, 0, stream>>>(
        GATES, CST, HALL, XH, s);
  }

  // --- output projection ---------------------------------------------------
  {
    int n = B_ * S_ * NC_;
    generator_kernel<<<(n + 255) / 256, 256, 0, stream>>>(HALL, Wgen, bgen, out);
  }
}
